// BioGPTRelationExtractor_39762807226990
// MI455X (gfx1250) — matrix operand order repaired and checked by reading, not compiled, not run on hardware
//
#include <hip/hip_runtime.h>
#include <hip/hip_bf16.h>
#include <math.h>

// ---------------------------------------------------------------------------
// Problem constants (from reference)
// ---------------------------------------------------------------------------
#define N_NODES 50000
#define D_IN    1024
#define HID     256
#define E_RAW   400000
#define E_TOT   450000          // + self loops
#define NEG_SLOPE 0.2f

// GEMM tiling
#define BM 128
#define BN 64
#define BK 32
#define LDSP 40                 // padded ushort row stride (multiple of 8 -> aligned b128)
#define A_BUF (BM * LDSP)
#define B_BUF (BN * LDSP)

typedef __attribute__((ext_vector_type(16))) __bf16        v16bf;
typedef __attribute__((ext_vector_type(8)))  float         v8f;
typedef __attribute__((ext_vector_type(8)))  unsigned int  v8u;
typedef unsigned short ushort_t;

// ---------------------------------------------------------------------------
// helpers
// ---------------------------------------------------------------------------
__device__ __forceinline__ void bf16split(float a, unsigned short& hi, unsigned short& lo) {
    // a ~= hi + lo  (hi = truncated bf16 (exact), lo = RNE bf16 of residual)
    unsigned u  = __float_as_uint(a);
    unsigned hu = u & 0xFFFF0000u;
    float    hf = __uint_as_float(hu);
    float    lf = a - hf;
    hi = (unsigned short)(hu >> 16);
    unsigned ul = __float_as_uint(lf);
    unsigned r  = ul + 0x7FFFu + ((ul >> 16) & 1u);   // round to nearest even
    lo = (unsigned short)(r >> 16);
}

// order-preserving float<->uint for atomicMax-based segment max
__device__ __forceinline__ unsigned enc_f32(float x) {
    unsigned u = __float_as_uint(x);
    return (u & 0x80000000u) ? ~u : (u | 0x80000000u);
}
__device__ __forceinline__ float dec_f32(unsigned u) {
    unsigned b = (u & 0x80000000u) ? (u & 0x7FFFFFFFu) : ~u;
    return __uint_as_float(b);
}
#define ENC_NEG_INF 0x007FFFFFu   // enc(-inf)

__device__ __forceinline__ void edge_sd(const int* __restrict__ ei, int e, int& s, int& d) {
    if (e < E_RAW) { s = ei[e]; d = ei[E_RAW + e]; }
    else           { s = e - E_RAW; d = s; }        // self loop
}

// ---------------------------------------------------------------------------
// fills (graph-capture safe replacement for memset)
// ---------------------------------------------------------------------------
__global__ void fill_f32(float* p, size_t n, float v) {
    size_t i = (size_t)blockIdx.x * blockDim.x + threadIdx.x;
    if (i < n) p[i] = v;
}
__global__ void fill_u32(unsigned* p, size_t n, unsigned v) {
    size_t i = (size_t)blockIdx.x * blockDim.x + threadIdx.x;
    if (i < n) p[i] = v;
}

// ---------------------------------------------------------------------------
// fp32 -> planar (hi,lo) bf16, elementwise (for A operands)
// ---------------------------------------------------------------------------
__global__ void split_planar(const float* __restrict__ in, ushort_t* __restrict__ hi,
                             ushort_t* __restrict__ lo, size_t n) {
    size_t i = (size_t)blockIdx.x * blockDim.x + threadIdx.x;
    if (i >= n) return;
    unsigned short h, l;
    bf16split(in[i], h, l);
    hi[i] = h; lo[i] = l;
}

// fp32 B[K,Nc] -> transposed planar bf16 Bt[Nc,K] (so B columns are K-contiguous)
__global__ void split_transpose(const float* __restrict__ B, ushort_t* __restrict__ hi,
                                ushort_t* __restrict__ lo, int K, int Nc) {
    int n = blockIdx.x * blockDim.x + threadIdx.x;
    int k = blockIdx.y;
    if (n >= Nc) return;
    unsigned short h, l;
    bf16split(B[(size_t)k * Nc + n], h, l);
    hi[(size_t)n * K + k] = h;
    lo[(size_t)n * K + k] = l;
}

// ---------------------------------------------------------------------------
// tile staging helpers (named regs only -> SROA, no scratch)
// ---------------------------------------------------------------------------
__device__ __forceinline__ void load_tile6(
    const ushort_t* __restrict__ Ahi, const ushort_t* __restrict__ Alo,
    const ushort_t* __restrict__ Bthi, const ushort_t* __restrict__ Btlo,
    int M, int K, int blockM, int blockN, int t, int kt,
    uint4& ah0, uint4& ah1, uint4& al0, uint4& al1, uint4& bh, uint4& bl)
{
    {   // A chunk 0  (128x32 ushorts/plane = 512 b128 chunks, 2/thread/plane)
        int row = t >> 2, kc = (t & 3) << 3;
        int gr = blockM + row; if (gr >= M) gr = M - 1;
        size_t off = (size_t)gr * K + kt + kc;
        ah0 = *(const uint4*)(Ahi + off);
        al0 = *(const uint4*)(Alo + off);
    }
    {   // A chunk 1
        int q = t + 256, row = q >> 2, kc = (q & 3) << 3;
        int gr = blockM + row; if (gr >= M) gr = M - 1;
        size_t off = (size_t)gr * K + kt + kc;
        ah1 = *(const uint4*)(Ahi + off);
        al1 = *(const uint4*)(Alo + off);
    }
    {   // B chunk (64x32 ushorts/plane = 256 chunks, 1/thread/plane)
        int n = t >> 2, kc = (t & 3) << 3;
        size_t off = (size_t)(blockN + n) * K + kt + kc;
        bh = *(const uint4*)(Bthi + off);
        bl = *(const uint4*)(Btlo + off);
    }
}

__device__ __forceinline__ void store_tile6(
    ushort_t* sAhi, ushort_t* sAlo, ushort_t* sBhi, ushort_t* sBlo, int t,
    const uint4& ah0, const uint4& ah1, const uint4& al0, const uint4& al1,
    const uint4& bh, const uint4& bl)
{
    {
        int row = t >> 2, kc = (t & 3) << 3;
        *(uint4*)(sAhi + row * LDSP + kc) = ah0;
        *(uint4*)(sAlo + row * LDSP + kc) = al0;
    }
    {
        int q = t + 256, row = q >> 2, kc = (q & 3) << 3;
        *(uint4*)(sAhi + row * LDSP + kc) = ah1;
        *(uint4*)(sAlo + row * LDSP + kc) = al1;
    }
    {
        int n = t >> 2, kc = (t & 3) << 3;
        *(uint4*)(sBhi + n * LDSP + kc) = bh;
        *(uint4*)(sBlo + n * LDSP + kc) = bl;
    }
}

// ---------------------------------------------------------------------------
// C[M,Nc] = A[M,K] @ B[K,Nc] + bias, from pre-split planar bf16 operands.
// A planes: [M,K] row-major.  B planes: pre-transposed [Nc,K].
// 256 threads = 8 waves, block tile 128x64; wave grid 4x2, 32x32 per wave
// (2x2 WMMA tiles, 3 split-product WMMAs each = 12 wmma / wave / K-step).
// LDS double-buffered; next tile's global loads are in flight across the
// whole WMMA phase; single barrier per K-step.
// ---------------------------------------------------------------------------
__global__ __launch_bounds__(256)
void gemm_bf16_planar(const ushort_t* __restrict__ Ahi, const ushort_t* __restrict__ Alo,
                      const ushort_t* __restrict__ Bthi, const ushort_t* __restrict__ Btlo,
                      const float* __restrict__ bias, float* __restrict__ C,
                      int M, int Nc, int K)
{
    __shared__ ushort_t sA_hi[2 * A_BUF];
    __shared__ ushort_t sA_lo[2 * A_BUF];
    __shared__ ushort_t sB_hi[2 * B_BUF];
    __shared__ ushort_t sB_lo[2 * B_BUF];

    const int t    = threadIdx.x;
    const int lane = t & 31;
    const int wave = t >> 5;
    const int l16  = lane & 15;
    const int half = lane >> 4;

    const int blockM = blockIdx.x * BM;
    const int blockN = blockIdx.y * BN;
    const int mbase  = (wave & 3) << 5;   // 0,32,64,96
    const int nbase  = (wave >> 2) << 5;  // 0,32

    v8f acc[2][2];
#pragma unroll
    for (int mi = 0; mi < 2; ++mi)
#pragma unroll
        for (int ni = 0; ni < 2; ++ni)
#pragma unroll
            for (int r = 0; r < 8; ++r) acc[mi][ni][r] = 0.0f;

    uint4 ah0, ah1, al0, al1, bh, bl;

    // prologue: fill buffer 0
    load_tile6(Ahi, Alo, Bthi, Btlo, M, K, blockM, blockN, t, 0,
               ah0, ah1, al0, al1, bh, bl);
    store_tile6(sA_hi, sA_lo, sB_hi, sB_lo, t, ah0, ah1, al0, al1, bh, bl);
    __syncthreads();

#pragma unroll 2
    for (int kt = 0; kt < K; kt += BK) {
        const int cur = (kt / BK) & 1;
        const int nxt = cur ^ 1;
        int ktn = kt + BK; if (ktn >= K) ktn = 0;   // clamped: branch-free pipeline

        // next tile's global loads -> in flight during the WMMA phase
        load_tile6(Ahi, Alo, Bthi, Btlo, M, K, blockM, blockN, t, ktn,
                   ah0, ah1, al0, al1, bh, bl);

        const ushort_t* cAhi = sA_hi + cur * A_BUF;
        const ushort_t* cAlo = sA_lo + cur * A_BUF;
        const ushort_t* cBhi = sB_hi + cur * B_BUF;
        const ushort_t* cBlo = sB_lo + cur * B_BUF;

        // A fragments (ISA 7.12.2 16-bit A layout): two 16-row strips
        v16bf fah[2], fal[2];
#pragma unroll
        for (int mi = 0; mi < 2; ++mi) {
            const int arow = mbase + mi * 16 + l16;
            v8u ahf, alf;
#pragma unroll
            for (int v = 0; v < 8; ++v) {
                int k = ((v >> 2) << 4) + (half << 3) + ((v & 3) << 1);
                ahf[v] = *(const unsigned*)(cAhi + arow * LDSP + k);
                alf[v] = *(const unsigned*)(cAlo + arow * LDSP + k);
            }
            fah[mi] = __builtin_bit_cast(v16bf, ahf);
            fal[mi] = __builtin_bit_cast(v16bf, alf);
        }

#pragma unroll
        for (int ni = 0; ni < 2; ++ni) {
            const int bcol = nbase + ni * 16 + l16;  // lane holds one output column
            v8u bhf, blf;
#pragma unroll
            for (int v = 0; v < 8; ++v) {
                int k = ((v >> 2) << 4) + (half << 3) + ((v & 3) << 1);  // same K order as A fragment
                bhf[v] = *(const unsigned*)(cBhi + bcol * LDSP + k);
                blf[v] = *(const unsigned*)(cBlo + bcol * LDSP + k);
            }
            const v16bf fbh = __builtin_bit_cast(v16bf, bhf);
            const v16bf fbl = __builtin_bit_cast(v16bf, blf);
#pragma unroll
            for (int mi = 0; mi < 2; ++mi) {
                // 3-term split product: hi*hi + lo*hi + hi*lo (~fp32 accuracy)
                acc[mi][ni] = __builtin_amdgcn_wmma_f32_16x16x32_bf16(false, fah[mi], false, fbh,
                                                                      (short)0, acc[mi][ni], false, false);
                acc[mi][ni] = __builtin_amdgcn_wmma_f32_16x16x32_bf16(false, fal[mi], false, fbh,
                                                                      (short)0, acc[mi][ni], false, false);
                acc[mi][ni] = __builtin_amdgcn_wmma_f32_16x16x32_bf16(false, fah[mi], false, fbl,
                                                                      (short)0, acc[mi][ni], false, false);
            }
        }

        // stage next tile into the other LDS buffer; one barrier per K-step
        store_tile6(sA_hi + nxt * A_BUF, sA_lo + nxt * A_BUF,
                    sB_hi + nxt * B_BUF, sB_lo + nxt * B_BUF, t,
                    ah0, ah1, al0, al1, bh, bl);
        __syncthreads();
    }

    // epilogue: C layout (VGPR r -> row half*8+r, lane l16 -> col) + bias
#pragma unroll
    for (int ni = 0; ni < 2; ++ni) {
        const int gcol = blockN + nbase + ni * 16 + l16;
        const float bv = bias[gcol];
#pragma unroll
        for (int mi = 0; mi < 2; ++mi)
#pragma unroll
            for (int r = 0; r < 8; ++r) {
                int grow = blockM + mbase + mi * 16 + half * 8 + r;
                if (grow < M) C[(size_t)grow * Nc + gcol] = acc[mi][ni][r] + bv;
            }
    }
}

// ---------------------------------------------------------------------------
// edge pass 1: alpha[e,h] = att_h . leakyrelu(xl[src]+xr[dst]); segment max
// one wave per edge
// ---------------------------------------------------------------------------
template <int H>
__global__ __launch_bounds__(256)
void edge_logits(const int* __restrict__ ei, const float* __restrict__ xl,
                 const float* __restrict__ xr, const float* __restrict__ att,
                 float* __restrict__ alpha, unsigned* __restrict__ amax_enc)
{
    const int e = blockIdx.x * 8 + (threadIdx.x >> 5);
    if (e >= E_TOT) return;
    const int lane = threadIdx.x & 31;
    int s, d; edge_sd(ei, e, s, d);
    const float* xls = xl + (size_t)s * (H * HID);
    const float* xrd = xr + (size_t)d * (H * HID);
#pragma unroll
    for (int h = 0; h < H; ++h) {
        float sum = 0.0f;
#pragma unroll
        for (int c = lane; c < HID; c += 32) {
            float m  = xls[h * HID + c] + xrd[h * HID + c];
            float lr = m > 0.0f ? m : NEG_SLOPE * m;
            sum += lr * att[h * HID + c];
        }
#pragma unroll
        for (int off = 16; off; off >>= 1) sum += __shfl_xor(sum, off, 32);
        if (lane == 0) {
            alpha[(size_t)e * H + h] = sum;
            atomicMax(&amax_enc[(size_t)d * H + h], enc_f32(sum));
        }
    }
}

// ---------------------------------------------------------------------------
// edge pass 2: w = exp(alpha - amax[dst]); denom[dst] += w; agg[dst] += w*xl[src]
// ---------------------------------------------------------------------------
template <int H>
__global__ __launch_bounds__(256)
void edge_accum(const int* __restrict__ ei, const float* __restrict__ xl,
                const float* __restrict__ alpha, const unsigned* __restrict__ amax_enc,
                float* __restrict__ denom, float* __restrict__ agg)
{
    const int e = blockIdx.x * 8 + (threadIdx.x >> 5);
    if (e >= E_TOT) return;
    const int lane = threadIdx.x & 31;
    int s, d; edge_sd(ei, e, s, d);
#pragma unroll
    for (int h = 0; h < H; ++h) {
        const float a  = alpha[(size_t)e * H + h];
        const float mx = dec_f32(amax_enc[(size_t)d * H + h]);
        const float w  = __expf(a - mx);
        if (lane == 0) atomicAdd(&denom[(size_t)d * H + h], w);
        const float* xs = xl  + (size_t)s * (H * HID) + h * HID;
        float*       ag = agg + (size_t)d * (H * HID) + h * HID;
#pragma unroll
        for (int c = lane; c < HID; c += 32)
            atomicAdd(&ag[c], w * xs[c]);
    }
}

// ---------------------------------------------------------------------------
// finalize: out = relu(agg / (denom + 1e-16) + bias)
// ---------------------------------------------------------------------------
template <int H>
__global__ void gat_finalize(const float* __restrict__ denom, const float* __restrict__ bias,
                             const float* __restrict__ agg, float* __restrict__ out)
{
    const size_t i = (size_t)blockIdx.x * blockDim.x + threadIdx.x;
    const size_t total = (size_t)N_NODES * H * HID;
    if (i >= total) return;
    const int node = (int)(i / (H * HID));
    const int hc   = (int)(i % (H * HID));
    const int h    = hc / HID;
    float v = agg[i] / (denom[(size_t)node * H + h] + 1e-16f) + bias[hc];
    out[i] = v > 0.0f ? v : 0.0f;
}

// ---------------------------------------------------------------------------
// host-side orchestration
// ---------------------------------------------------------------------------
extern "C" void kernel_launch(void* const* d_in, const int* in_sizes, int n_in,
                              void* d_out, int out_size, void* d_ws, size_t ws_size,
                              hipStream_t stream)
{
    (void)in_sizes; (void)n_in; (void)out_size; (void)ws_size;

    const float* X     = (const float*)d_in[0];
    const int*   EI    = (const int*)  d_in[1];
    const float* W1l   = (const float*)d_in[2];
    const float* b1l   = (const float*)d_in[3];
    const float* W1r   = (const float*)d_in[4];
    const float* b1r   = (const float*)d_in[5];
    const float* att1  = (const float*)d_in[6];
    const float* bias1 = (const float*)d_in[7];
    const float* W2l   = (const float*)d_in[8];
    const float* b2l   = (const float*)d_in[9];
    const float* W2r   = (const float*)d_in[10];
    const float* b2r   = (const float*)d_in[11];
    const float* att2  = (const float*)d_in[12];
    const float* bias2 = (const float*)d_in[13];
    float* out = (float*)d_out;

    const size_t NF1 = (size_t)N_NODES * 1024;   // 4 heads * 256
    const size_t NF2 = (size_t)N_NODES * 256;    // 1 head  * 256

    // ---- workspace layout (floats / ushorts)
    float*    xl1    = (float*)d_ws;
    float*    xr1    = xl1 + NF1;
    float*    agg1   = xr1 + NF1;                 // becomes x2 after finalize (in place)
    float*    alpha1 = agg1 + NF1;                // E_TOT*4 floats
    unsigned* amax1  = (unsigned*)(alpha1 + (size_t)E_TOT * 4);
    float*    denom1 = (float*)(amax1 + (size_t)N_NODES * 4);
    ushort_t* pAhi   = (ushort_t*)(denom1 + (size_t)N_NODES * 4);  // planar bf16 A (X / x2)
    ushort_t* pAlo   = pAhi + NF1;
    ushort_t* pWAhi  = pAlo + NF1;                // W (left) planes, max 1024x1024
    ushort_t* pWAlo  = pWAhi + (size_t)1024 * 1024;
    ushort_t* pWBhi  = pWAlo + (size_t)1024 * 1024;   // W (right) planes
    ushort_t* pWBlo  = pWBhi + (size_t)1024 * 1024;
    // ---- layer-2 buffers alias the (then-dead) xl1/xr1 region
    float*    xl2    = xl1;
    float*    xr2    = xl1 + NF2;
    float*    agg2   = xl1 + 2 * NF2;
    float*    alpha2 = xl1 + 3 * NF2;
    unsigned* amax2  = (unsigned*)(alpha2 + (size_t)E_TOT);
    float*    denom2 = (float*)(amax2 + (size_t)N_NODES);

    const dim3 blk(256);
    const int  gM      = (N_NODES + BM - 1) / BM;           // 391
    const int  eBlocks = (E_TOT + 7) / 8;                   // 1 wave / edge
    auto gfill = [](size_t n) { return (unsigned)((n + 255) / 256); };

    // ================= layer 1: GATv2Conv(1024 -> 256, heads=4) =================
    split_planar<<<gfill(NF1), blk, 0, stream>>>(X, pAhi, pAlo, NF1);
    split_transpose<<<dim3(1024 / 256, 1024), blk, 0, stream>>>(W1l, pWAhi, pWAlo, 1024, 1024);
    split_transpose<<<dim3(1024 / 256, 1024), blk, 0, stream>>>(W1r, pWBhi, pWBlo, 1024, 1024);

    fill_u32<<<gfill((size_t)N_NODES * 4), blk, 0, stream>>>(amax1, (size_t)N_NODES * 4, ENC_NEG_INF);
    fill_f32<<<gfill((size_t)N_NODES * 4), blk, 0, stream>>>(denom1, (size_t)N_NODES * 4, 0.0f);
    fill_f32<<<gfill(NF1), blk, 0, stream>>>(agg1, NF1, 0.0f);

    gemm_bf16_planar<<<dim3(gM, 1024 / BN), blk, 0, stream>>>(pAhi, pAlo, pWAhi, pWAlo, b1l, xl1,
                                                              N_NODES, 1024, D_IN);
    gemm_bf16_planar<<<dim3(gM, 1024 / BN), blk, 0, stream>>>(pAhi, pAlo, pWBhi, pWBlo, b1r, xr1,
                                                              N_NODES, 1024, D_IN);

    edge_logits<4><<<eBlocks, blk, 0, stream>>>(EI, xl1, xr1, att1, alpha1, amax1);
    edge_accum<4><<<eBlocks, blk, 0, stream>>>(EI, xl1, alpha1, amax1, denom1, agg1);
    gat_finalize<4><<<gfill(NF1), blk, 0, stream>>>(denom1, bias1, agg1, agg1);   // -> x2 (relu'd)

    // ================= layer 2: GATv2Conv(1024 -> 256, heads=1) =================
    split_planar<<<gfill(NF1), blk, 0, stream>>>(agg1, pAhi, pAlo, NF1);          // x2 planes
    split_transpose<<<dim3(1, 1024), blk, 0, stream>>>(W2l, pWAhi, pWAlo, 1024, 256);
    split_transpose<<<dim3(1, 1024), blk, 0, stream>>>(W2r, pWBhi, pWBlo, 1024, 256);

    fill_u32<<<gfill((size_t)N_NODES), blk, 0, stream>>>(amax2, (size_t)N_NODES, ENC_NEG_INF);
    fill_f32<<<gfill((size_t)N_NODES), blk, 0, stream>>>(denom2, (size_t)N_NODES, 0.0f);
    fill_f32<<<gfill(NF2), blk, 0, stream>>>(agg2, NF2, 0.0f);

    gemm_bf16_planar<<<dim3(gM, 256 / BN), blk, 0, stream>>>(pAhi, pAlo, pWAhi, pWAlo, b2l, xl2,
                                                             N_NODES, 256, 1024);
    gemm_bf16_planar<<<dim3(gM, 256 / BN), blk, 0, stream>>>(pAhi, pAlo, pWBhi, pWBlo, b2r, xr2,
                                                             N_NODES, 256, 1024);

    edge_logits<1><<<eBlocks, blk, 0, stream>>>(EI, xl2, xr2, att2, alpha2, amax2);
    edge_accum<1><<<eBlocks, blk, 0, stream>>>(EI, xl2, alpha2, amax2, denom2, agg2);
    gat_finalize<1><<<gfill(NF2), blk, 0, stream>>>(denom2, bias2, agg2, out);
}
